// VAE_RNN_PHYNN_5033701671482
// MI455X (gfx1250) — matrix operand order repaired and checked by reading, not compiled, not run on hardware
//
#include <hip/hip_runtime.h>
#include <hip/hip_bf16.h>

typedef __bf16 bf16_t;
typedef __attribute__((ext_vector_type(16))) __bf16 v16bf;
typedef __attribute__((ext_vector_type(8)))  float  v8f;
typedef __attribute__((ext_vector_type(4)))  unsigned int u32x4;

#define LOG2PI 1.8378770664093453f
#define MPNT_WT 1.0f

union FragCvt { u32x4 q[2]; v16bf v; };

// A-matrix 16x32 bf16 fragment (ISA 7.12.2): lanes 0-15 hold K={kb..kb+7, kb+16..kb+23},
// lanes 16-31 hold K={kb+8..kb+15, kb+24..kb+31}. Caller pre-offsets by (lane>>4)*8.
__device__ inline v16bf ld_afrag(const bf16_t* p) {
  FragCvt u;
  u.q[0] = *(const u32x4*)(p);
  u.q[1] = *(const u32x4*)(p + 16);
  return u.v;
}
// B-matrix 32x16 bf16 fragment: lanes 0-15 hold K={kb..kb+7, kb+16..kb+23} of row N=lane&15,
// lanes 16-31 hold K={kb+8..kb+15, kb+24..kb+31}. Caller pre-offsets by (lane>>4)*8.
__device__ inline v16bf ld_bfrag(const bf16_t* p) {
  FragCvt u;
  u.q[0] = *(const u32x4*)(p);
  u.q[1] = *(const u32x4*)(p + 16);
  return u.v;
}

// O[16 x 16] tile (ntile) of  X[16 x K] @ W[N x K]^T  accumulated into acc.
__device__ inline v8f wmma_acc(const bf16_t* A, int lda, const bf16_t* W, int ldw,
                               int K, int ntile, int lane, v8f acc) {
  const int r = lane & 15;
  const bf16_t* ap = A + r * lda + ((lane >> 4) << 3);
  const bf16_t* bp = W + (size_t)(ntile * 16 + r) * ldw + ((lane >> 4) << 3);
#pragma unroll 4
  for (int kb = 0; kb < K; kb += 32) {
    v16bf a = ld_afrag(ap + kb);
    v16bf b = ld_bfrag(bp + kb);
    acc = __builtin_amdgcn_wmma_f32_16x16x32_bf16(false, a, false, b,
                                                  (short)0, acc, false, false);
  }
  return acc;
}

// C/D layout: elem i of v8f -> M = i + (lane>>4)*8, N = ntile*16 + (lane&15).
__device__ inline void store_out(v8f acc, const float* bias, int nvalid,
                                 float* fbuf, int ldf, bf16_t* bbuf, int ldb,
                                 int ntile, int lane, bool relu) {
  const int col = ntile * 16 + (lane & 15);
  const int mb = (lane >> 4) << 3;
  float bv = (bias && col < nvalid) ? bias[col] : 0.0f;
#pragma unroll
  for (int i = 0; i < 8; ++i) {
    int m = mb + i;
    float v = acc[i] + bv;
    if (relu) v = fmaxf(v, 0.0f);
    if (fbuf) fbuf[m * ldf + col] = v;
    if (bbuf) bbuf[m * ldb + col] = (bf16_t)v;
  }
}

struct KParams {
  const float *u, *y, *eps;
  const bf16_t *phiy1, *phiy2, *phiu1, *phiu2, *phiz1, *phiz2;
  const bf16_t *enc1, *enc2, *encm, *enclv;
  const bf16_t *pri1, *pri2, *prim, *prilv;
  const bf16_t *dcw1, *dcw2, *dcm, *dclv;
  const bf16_t *wih, *whh, *Ap, *Bp, *Cp;
  const float *b_phiy1, *b_phiy2, *b_phiu1, *b_phiu2, *b_phiz1, *b_phiz2;
  const float *b_enc1, *b_enc2, *b_encm, *b_enclv;
  const float *b_pri1, *b_pri2, *b_prim, *b_prilv;
  const float *b_dec1, *b_dec2, *b_dcm, *b_dclv;
  const float *Cf;
  const float *sigma;
  float *out;
};

__global__ void cvt_pad_kernel(const float* src, bf16_t* dst, int N, int K, int Np, int Kp) {
  int i = blockIdx.x * blockDim.x + threadIdx.x;
  int tot = Np * Kp;
  if (i >= tot) return;
  int n = i / Kp, k = i % Kp;
  float v = (n < N && k < K) ? src[n * K + k] : 0.0f;
  dst[i] = (bf16_t)v;
}

__global__ void zero_out_kernel(float* p) {
  if (threadIdx.x == 0 && blockIdx.x == 0) *p = 0.0f;
}

__global__ __launch_bounds__(256) void vrnn_kernel(KParams P) {
  constexpr int T = 1024, H = 128, Z = 16, Y = 8, U = 8;

  __shared__ __align__(16) bf16_t s_cat[16 * 256];   // [:,0:128]=phi_y  [:,128:256]=h(bf16)
  __shared__ __align__(16) bf16_t s_phiu[16 * 128];
  __shared__ __align__(16) bf16_t s_tmp [16 * 128];
  __shared__ __align__(16) bf16_t s_tmp2[16 * 128];
  __shared__ __align__(16) bf16_t s_ench[16 * 128];
  __shared__ __align__(16) bf16_t s_pr  [16 * 128];
  __shared__ __align__(16) bf16_t s_phiz[16 * 128];
  __shared__ __align__(16) bf16_t s_zpad[16 * 32];   // z_t, K-padded to 32
  __shared__ __align__(16) bf16_t s_ypad[16 * 32];
  __shared__ __align__(16) bf16_t s_upad[16 * 32];
  __shared__ __align__(16) bf16_t s_uprev[16 * 32];
  __shared__ float s_h[16 * 128];
  __shared__ float s_yt[16 * 8];
  __shared__ float s_emean[256], s_elv[256], s_zphy[256];
  __shared__ float s_pm[256], s_plv[256], s_ev[256];
  __shared__ float s_dm[256], s_dlv[256], s_mmean[256];
  __shared__ float s_C[8 * 16];

  const int tid = threadIdx.x;
  const int wv = tid >> 5;
  const int lane = tid & 31;
  const int b0 = blockIdx.x * 16;

  for (int i = tid; i < 16 * 256; i += 256) s_cat[i] = (bf16_t)0.0f;
  for (int i = tid; i < 16 * 32; i += 256) {
    s_zpad[i] = (bf16_t)0.0f; s_ypad[i] = (bf16_t)0.0f;
    s_upad[i] = (bf16_t)0.0f; s_uprev[i] = (bf16_t)0.0f;
  }
  for (int i = tid; i < 16 * 128; i += 256) s_h[i] = 0.0f;
  if (tid < 128) s_C[tid] = P.Cf[tid];
  const float sg = P.sigma[0];
  const float sig2 = sg * sg;
  float lloss = 0.0f;
  __syncthreads();

  for (int t = 0; t < T; ++t) {
    // ---- P0: stage inputs ----
    if (tid < 128) {
      int m = tid >> 3, j = tid & 7;
      float yv = P.y[(size_t)(b0 + m) * Y * T + (size_t)j * T + t];
      s_yt[m * 8 + j] = yv;
      s_ypad[m * 32 + j] = (bf16_t)yv;
      float uv = P.u[(size_t)(b0 + m) * U * T + (size_t)j * T + t];
      s_upad[m * 32 + j] = (bf16_t)uv;
    } else {
      int q = tid - 128; int m = q >> 3, j = q & 7;
      float up = (t == 0) ? 0.0f
                          : P.u[(size_t)(b0 + m) * U * T + (size_t)j * T + (t - 1)];
      s_uprev[m * 32 + j] = (bf16_t)up;
    }
    __syncthreads();

    // ---- P1: gh = h @ Whh^T (regs), phi_y L1, phi_u L1 ----
    v8f ghr = {}, ghz = {}, ghn = {};
    ghr = wmma_acc(s_cat + 128, 256, P.whh, 128, 128, wv,      lane, ghr);
    ghz = wmma_acc(s_cat + 128, 256, P.whh, 128, 128, wv + 8,  lane, ghz);
    ghn = wmma_acc(s_cat + 128, 256, P.whh, 128, 128, wv + 16, lane, ghn);
    { v8f c = {}; c = wmma_acc(s_ypad, 32, P.phiy1, 32, 32, wv, lane, c);
      store_out(c, P.b_phiy1, 128, nullptr, 0, s_tmp, 128, wv, lane, true); }
    { v8f c = {}; c = wmma_acc(s_upad, 32, P.phiu1, 32, 32, wv, lane, c);
      store_out(c, P.b_phiu1, 128, nullptr, 0, s_tmp2, 128, wv, lane, true); }
    __syncthreads();

    // ---- P2: phi_y L2 -> cat[:, :128], phi_u L2 ----
    { v8f c = {}; c = wmma_acc(s_tmp, 128, P.phiy2, 128, 128, wv, lane, c);
      store_out(c, P.b_phiy2, 128, nullptr, 0, s_cat, 256, wv, lane, false); }
    { v8f c = {}; c = wmma_acc(s_tmp2, 128, P.phiu2, 128, 128, wv, lane, c);
      store_out(c, P.b_phiu2, 128, nullptr, 0, s_phiu, 128, wv, lane, false); }
    __syncthreads();

    // ---- P3: gi = phi_u @ Wih^T (regs), prior L1, enc L1 (K=256 over [phi_y|h]) ----
    v8f gir = {}, giz = {}, gin = {};
    gir = wmma_acc(s_phiu, 128, P.wih, 128, 128, wv,      lane, gir);
    giz = wmma_acc(s_phiu, 128, P.wih, 128, 128, wv + 8,  lane, giz);
    gin = wmma_acc(s_phiu, 128, P.wih, 128, 128, wv + 16, lane, gin);
    { v8f c = {}; c = wmma_acc(s_cat + 128, 256, P.pri1, 128, 128, wv, lane, c);
      store_out(c, P.b_pri1, 128, nullptr, 0, s_tmp, 128, wv, lane, true); }
    { v8f c = {}; c = wmma_acc(s_cat, 256, P.enc1, 256, 256, wv, lane, c);
      store_out(c, P.b_enc1, 128, nullptr, 0, s_tmp2, 128, wv, lane, true); }
    __syncthreads();

    // ---- P4: prior L2, enc L2 ----
    { v8f c = {}; c = wmma_acc(s_tmp, 128, P.pri2, 128, 128, wv, lane, c);
      store_out(c, P.b_pri2, 128, nullptr, 0, s_pr, 128, wv, lane, true); }
    { v8f c = {}; c = wmma_acc(s_tmp2, 128, P.enc2, 128, 128, wv, lane, c);
      store_out(c, P.b_enc2, 128, nullptr, 0, s_ench, 128, wv, lane, true); }
    __syncthreads();

    // ---- P5: heads (one wave each) ----
    if (wv == 0) {
      v8f c = {}; c = wmma_acc(s_ench, 128, P.encm, 128, 128, 0, lane, c);
      store_out(c, P.b_encm, 16, s_emean, 16, nullptr, 0, 0, lane, false);
    } else if (wv == 1) {
      v8f c = {}; c = wmma_acc(s_ench, 128, P.enclv, 128, 128, 0, lane, c);
      store_out(c, P.b_enclv, 16, s_elv, 16, nullptr, 0, 0, lane, false);
    } else if (wv == 2) {
      v8f c = {};  // z_phy = z_prev @ A^T + u_prev @ B^T  (s_zpad still holds z_{t-1})
      c = wmma_acc(s_zpad, 32, P.Ap, 32, 32, 0, lane, c);
      c = wmma_acc(s_uprev, 32, P.Bp, 32, 32, 0, lane, c);
      store_out(c, nullptr, 16, s_zphy, 16, nullptr, 0, 0, lane, false);
    } else if (wv == 3) {
      v8f c = {}; c = wmma_acc(s_pr, 128, P.prim, 128, 128, 0, lane, c);
      store_out(c, P.b_prim, 16, s_pm, 16, nullptr, 0, 0, lane, false);
    } else if (wv == 4) {
      v8f c = {}; c = wmma_acc(s_pr, 128, P.prilv, 128, 128, 0, lane, c);
      store_out(c, P.b_prilv, 16, s_plv, 16, nullptr, 0, 0, lane, false);
    }
    __syncthreads();

    // ---- P6: rsample z_t ----
    {
      int m = tid >> 4, k = tid & 15;
      float em = s_emean[tid] + s_zphy[tid];
      float elv = s_elv[tid];
      float ev = expf(elv);
      float ep = P.eps[(size_t)(b0 + m) * Z * T + (size_t)k * T + t];
      float z = em + sqrtf(ev) * ep;
      s_emean[tid] = em;
      s_ev[tid] = ev;
      s_zpad[m * 32 + k] = (bf16_t)z;
    }
    __syncthreads();

    // ---- P7..P10: phi_z trunk, dec trunk ----
    { v8f c = {}; c = wmma_acc(s_zpad, 32, P.phiz1, 32, 32, wv, lane, c);
      store_out(c, P.b_phiz1, 128, nullptr, 0, s_tmp, 128, wv, lane, true); }
    __syncthreads();
    { v8f c = {}; c = wmma_acc(s_tmp, 128, P.phiz2, 128, 128, wv, lane, c);
      store_out(c, P.b_phiz2, 128, nullptr, 0, s_phiz, 128, wv, lane, false); }
    __syncthreads();
    { v8f c = {}; c = wmma_acc(s_phiz, 128, P.dcw1, 128, 128, wv, lane, c);
      store_out(c, P.b_dec1, 128, nullptr, 0, s_tmp, 128, wv, lane, true); }
    __syncthreads();
    { v8f c = {}; c = wmma_acc(s_tmp, 128, P.dcw2, 128, 128, wv, lane, c);
      store_out(c, P.b_dec2, 128, nullptr, 0, s_tmp2, 128, wv, lane, true); }
    __syncthreads();

    // ---- P11: dec heads + mmean = z_t @ C^T ----
    if (wv == 0) {
      v8f c = {}; c = wmma_acc(s_tmp2, 128, P.dcm, 128, 128, 0, lane, c);
      store_out(c, P.b_dcm, 8, s_dm, 16, nullptr, 0, 0, lane, false);
    } else if (wv == 1) {
      v8f c = {}; c = wmma_acc(s_tmp2, 128, P.dclv, 128, 128, 0, lane, c);
      store_out(c, P.b_dclv, 8, s_dlv, 16, nullptr, 0, 0, lane, false);
    } else if (wv == 2) {
      v8f c = {}; c = wmma_acc(s_zpad, 32, P.Cp, 32, 32, 0, lane, c);
      store_out(c, nullptr, 16, s_mmean, 16, nullptr, 0, 0, lane, false);
    }
    __syncthreads();

    // ---- P12: losses (thread-local accumulate) + GRU combine ----
    {   // KLD: 256 terms
      float em = s_emean[tid], elv = s_elv[tid], pm = s_pm[tid], plv = s_plv[tid];
      float d = em - pm;
      lloss += 0.5f * (plv - elv - 1.0f + (s_ev[tid] + d * d) * expf(-plv));
    }
    if (tid < 128) {  // -loss_pred: 128 terms
      int m = tid >> 3, j = tid & 7;
      float yv = s_yt[m * 8 + j], dm = s_dm[m * 16 + j], dlv = s_dlv[m * 16 + j];
      float dd = yv - dm;
      lloss -= (-0.5f * LOG2PI - 0.5f * dlv - 0.5f * dd * dd * expf(-dlv));
    }
#pragma unroll
    for (int q = 0; q < 4; ++q) {  // -pen: 1024 terms, 4 per thread
      int flat = tid * 4 + q;
      int m = flat >> 6, i = (flat >> 3) & 7, j = flat & 7;
      float zv = 0.0f;
#pragma unroll 4
      for (int k = 0; k < 16; ++k)
        zv += s_C[i * 16 + k] * s_C[j * 16 + k] * s_ev[m * 16 + k];
      float ms2 = zv + sig2;
      float dd = s_yt[m * 8 + j] - s_mmean[m * 16 + j];
      lloss -= MPNT_WT * (-0.5f * LOG2PI - 0.5f * logf(ms2) - 0.5f * dd * dd / ms2);
    }
    {   // GRU: wave wv owns columns [16wv, 16wv+16)
      int col = wv * 16 + (lane & 15);
      int mb = (lane >> 4) << 3;
#pragma unroll
      for (int i = 0; i < 8; ++i) {
        int m = mb + i;
        float r  = 1.0f / (1.0f + expf(-(gir[i] + ghr[i])));
        float zt = 1.0f / (1.0f + expf(-(giz[i] + ghz[i])));
        float nn = tanhf(gin[i] + r * ghn[i]);
        float hv = s_h[m * H + col];
        float hn = (1.0f - zt) * nn + zt * hv;
        s_h[m * H + col] = hn;
        s_cat[m * 256 + 128 + col] = (bf16_t)hn;
      }
    }
    __syncthreads();
  }

  atomicAdd(P.out, lloss);
}

extern "C" void kernel_launch(void* const* d_in, const int* in_sizes, int n_in,
                              void* d_out, int out_size, void* d_ws, size_t ws_size,
                              hipStream_t stream) {
  (void)in_sizes; (void)n_in; (void)out_size; (void)ws_size;
  auto F = [&](int i) { return (const float*)d_in[i]; };

  bf16_t* w = (bf16_t*)d_ws;
  size_t off = 0;
  auto alloc = [&](size_t n) { bf16_t* p = w + off; off += n; return p; };

  bf16_t* phiy1 = alloc(128 * 32);  bf16_t* phiy2 = alloc(128 * 128);
  bf16_t* phiu1 = alloc(128 * 32);  bf16_t* phiu2 = alloc(128 * 128);
  bf16_t* phiz1 = alloc(128 * 32);  bf16_t* phiz2 = alloc(128 * 128);
  bf16_t* enc1  = alloc(128 * 256); bf16_t* enc2  = alloc(128 * 128);
  bf16_t* encm  = alloc(16 * 128);  bf16_t* enclv = alloc(16 * 128);
  bf16_t* pri1  = alloc(128 * 128); bf16_t* pri2  = alloc(128 * 128);
  bf16_t* prim  = alloc(16 * 128);  bf16_t* prilv = alloc(16 * 128);
  bf16_t* dcw1  = alloc(128 * 128); bf16_t* dcw2  = alloc(128 * 128);
  bf16_t* dcm   = alloc(16 * 128);  bf16_t* dclv  = alloc(16 * 128);
  bf16_t* wih   = alloc(384 * 128); bf16_t* whh   = alloc(384 * 128);
  bf16_t* Ap    = alloc(16 * 32);   bf16_t* Bp    = alloc(16 * 32);
  bf16_t* Cp    = alloc(16 * 32);

  auto cvt = [&](const float* s, bf16_t* d, int N, int K, int Np, int Kp) {
    int tot = Np * Kp;
    cvt_pad_kernel<<<(tot + 255) / 256, 256, 0, stream>>>(s, d, N, K, Np, Kp);
  };
  cvt(F(3),  phiy1, 128, 8,   128, 32);
  cvt(F(5),  phiy2, 128, 128, 128, 128);
  cvt(F(7),  phiu1, 128, 8,   128, 32);
  cvt(F(9),  phiu2, 128, 128, 128, 128);
  cvt(F(11), phiz1, 128, 16,  128, 32);
  cvt(F(13), phiz2, 128, 128, 128, 128);
  cvt(F(15), enc1,  128, 256, 128, 256);
  cvt(F(17), enc2,  128, 128, 128, 128);
  cvt(F(19), encm,  16, 128,  16, 128);
  cvt(F(21), enclv, 16, 128,  16, 128);
  cvt(F(23), pri1,  128, 128, 128, 128);
  cvt(F(25), pri2,  128, 128, 128, 128);
  cvt(F(27), prim,  16, 128,  16, 128);
  cvt(F(29), prilv, 16, 128,  16, 128);
  cvt(F(31), dcw1,  128, 128, 128, 128);
  cvt(F(33), dcw2,  128, 128, 128, 128);
  cvt(F(35), dcm,   8, 128,   16, 128);
  cvt(F(37), dclv,  8, 128,   16, 128);
  cvt(F(39), wih,   384, 128, 384, 128);
  cvt(F(40), whh,   384, 128, 384, 128);
  cvt(F(41), Ap,    16, 16,   16, 32);
  cvt(F(42), Bp,    16, 8,    16, 32);
  cvt(F(43), Cp,    8, 16,    16, 32);

  zero_out_kernel<<<1, 1, 0, stream>>>((float*)d_out);

  KParams P;
  P.u = F(0); P.y = F(1); P.eps = F(2);
  P.phiy1 = phiy1; P.phiy2 = phiy2; P.phiu1 = phiu1; P.phiu2 = phiu2;
  P.phiz1 = phiz1; P.phiz2 = phiz2;
  P.enc1 = enc1; P.enc2 = enc2; P.encm = encm; P.enclv = enclv;
  P.pri1 = pri1; P.pri2 = pri2; P.prim = prim; P.prilv = prilv;
  P.dcw1 = dcw1; P.dcw2 = dcw2; P.dcm = dcm; P.dclv = dclv;
  P.wih = wih; P.whh = whh; P.Ap = Ap; P.Bp = Bp; P.Cp = Cp;
  P.b_phiy1 = F(4);  P.b_phiy2 = F(6);
  P.b_phiu1 = F(8);  P.b_phiu2 = F(10);
  P.b_phiz1 = F(12); P.b_phiz2 = F(14);
  P.b_enc1 = F(16);  P.b_enc2 = F(18);
  P.b_encm = F(20);  P.b_enclv = F(22);
  P.b_pri1 = F(24);  P.b_pri2 = F(26);
  P.b_prim = F(28);  P.b_prilv = F(30);
  P.b_dec1 = F(32);  P.b_dec2 = F(34);
  P.b_dcm = F(36);   P.b_dclv = F(38);
  P.Cf = F(43);
  P.sigma = F(44);
  P.out = (float*)d_out;

  vrnn_kernel<<<16, 256, 0, stream>>>(P);
}
